// TransformerBlock_68736656605711
// MI455X (gfx1250) — matrix operand order repaired and checked by reading, not compiled, not run on hardware
//
#include <hip/hip_runtime.h>
#include <hip/hip_bf16.h>

typedef __attribute__((ext_vector_type(16))) __bf16 v16bf;
typedef __attribute__((ext_vector_type(8)))  float  v8f;
typedef int v4i_ __attribute__((vector_size(16)));   // matches async-LDS builtin param

#define B_   4
#define S_   1024
#define D_   1024
#define H_   16
#define HD_  64
#define ROWS_ (B_ * S_)   // 4096

#if __has_builtin(__builtin_amdgcn_global_load_async_to_lds_b128) && \
    __has_builtin(__builtin_amdgcn_s_wait_asynccnt)
#define USE_ASYNC 1
#else
#define USE_ASYNC 0
#endif

// ---------------------------------------------------------------------------
// Transposing f32 -> bf16 weight conversion: in[K][N] -> out[N][K]
// LDS-tiled 64x64, coalesced both sides.
// ---------------------------------------------------------------------------
__global__ __launch_bounds__(256) void cvt_transpose(const float* __restrict__ in,
                                                     __bf16* __restrict__ out,
                                                     int K, int N) {
    __shared__ __bf16 tl[64 * 72];
    const int n0 = blockIdx.x * 64;
    const int k0 = blockIdx.y * 64;
    const int tid = threadIdx.x;
    #pragma unroll
    for (int it = 0; it < 4; ++it) {
        int idx = tid + it * 256;          // 1024 float4 chunks
        int r = idx >> 4;                  // k row in tile (0..63)
        int c = (idx & 15) * 4;            // n col
        float4 d = *(const float4*)(in + (size_t)(k0 + r) * N + n0 + c);
        tl[r * 72 + c + 0] = (__bf16)d.x;
        tl[r * 72 + c + 1] = (__bf16)d.y;
        tl[r * 72 + c + 2] = (__bf16)d.z;
        tl[r * 72 + c + 3] = (__bf16)d.w;
    }
    __syncthreads();
    {
        int n  = tid >> 2;                 // 0..63
        int kc = (tid & 3) * 16;
        __attribute__((aligned(16))) __bf16 tmp[16];
        #pragma unroll
        for (int j = 0; j < 16; ++j) tmp[j] = tl[(kc + j) * 72 + n];
        __bf16* dst = out + (size_t)(n0 + n) * K + k0 + kc;
        *(uint4*)dst       = ((uint4*)tmp)[0];
        *(uint4*)(dst + 8) = ((uint4*)tmp)[1];
    }
}

// ---------------------------------------------------------------------------
// LayerNorm (f32 in) -> bf16 out.  One block (256 thr) per row of D_ elems.
// ---------------------------------------------------------------------------
__global__ __launch_bounds__(256) void ln_bf16(const float* __restrict__ x,
                                               const float* __restrict__ sc,
                                               const float* __restrict__ sh,
                                               __bf16* __restrict__ out) {
    __shared__ float ws1[8], ws2[8];
    const int row = blockIdx.x;
    const float* xr = x + (size_t)row * D_;
    float s = 0.f, s2 = 0.f;
    for (int i = threadIdx.x; i < D_; i += 256) {
        float v = xr[i];
        s += v; s2 += v * v;
    }
    #pragma unroll
    for (int off = 16; off >= 1; off >>= 1) {
        s  += __shfl_xor(s,  off, 32);
        s2 += __shfl_xor(s2, off, 32);
    }
    const int wave = threadIdx.x >> 5;
    if ((threadIdx.x & 31) == 0) { ws1[wave] = s; ws2[wave] = s2; }
    __syncthreads();
    s = 0.f; s2 = 0.f;
    #pragma unroll
    for (int w = 0; w < 8; ++w) { s += ws1[w]; s2 += ws2[w]; }
    const float mean = s * (1.0f / D_);
    const float var  = s2 * (1.0f / D_) - mean * mean;
    const float rstd = rsqrtf(var + 1e-5f);
    for (int i = threadIdx.x; i < D_; i += 256)
        out[(size_t)row * D_ + i] = (__bf16)(sc[i] * ((xr[i] - mean) * rstd) + sh[i]);
}

// ---------------------------------------------------------------------------
// bf16 WMMA GEMM:  C[M,N] = A[M,K] * Wt[N,K]^T  (+ epilogue)
//   EP=0 : store bf16 raw
//   EP=1 : y = gelu(acc + bias[col]) -> bf16
//   EP=2 : y = acc + bias[col] + res[row,col] -> f32
// Block tile 128x128, 8 waves (4x2), wave tile 32x64, K-step 32.
// Double-buffered LDS; async-to-LDS pipeline when the toolchain has it.
// ---------------------------------------------------------------------------
template <int EP>
__global__ __launch_bounds__(256) void gemm_bf16(const __bf16* __restrict__ A,
                                                 const __bf16* __restrict__ Wt,
                                                 const float* __restrict__ bias,
                                                 const float* __restrict__ res,
                                                 void* __restrict__ outv,
                                                 int N, int K) {
    __shared__ __bf16 Asm[2][128 * 40];   // A tile  [row][k], stride 40
    __shared__ __bf16 Bsm[2][128 * 40];   // Wt tile [n][k],  stride 40

    const int tid  = threadIdx.x;
    const int wave = tid >> 5;
    const int lane = tid & 31;
    const int half = lane >> 4;
    const int l16  = lane & 15;
    const int row0 = blockIdx.y * 128;
    const int n0   = blockIdx.x * 128;
    const int wm   = (wave >> 1) * 32;
    const int wn   = (wave & 1) * 64;

    // cooperative loader mapping: 512 x 16B chunks per tile, 2 per thread
    const int lr = tid >> 2;              // 0..63  (+64 on second chunk)
    const int lk = (tid & 3) * 8;

    v8f acc[8] = {};

#if USE_ASYNC
    #pragma unroll
    for (int it = 0; it < 2; ++it) {
        int r = lr + it * 64;
        __builtin_amdgcn_global_load_async_to_lds_b128(
            (v4i_*)(A + (size_t)(row0 + r) * K + lk),
            (v4i_*)&Asm[0][r * 40 + lk], 0, 0);
        __builtin_amdgcn_global_load_async_to_lds_b128(
            (v4i_*)(Wt + (size_t)(n0 + r) * K + lk),
            (v4i_*)&Bsm[0][r * 40 + lk], 0, 0);
    }
    __builtin_amdgcn_s_wait_asynccnt(0);
#else
    uint4 ar[2], br[2];
    #pragma unroll
    for (int it = 0; it < 2; ++it) {
        int r = lr + it * 64;
        ar[it] = *(const uint4*)(A  + (size_t)(row0 + r) * K + lk);
        br[it] = *(const uint4*)(Wt + (size_t)(n0 + r) * K + lk);
    }
    #pragma unroll
    for (int it = 0; it < 2; ++it) {
        int r = lr + it * 64;
        *(uint4*)&Asm[0][r * 40 + lk] = ar[it];
        *(uint4*)&Bsm[0][r * 40 + lk] = br[it];
    }
#endif
    __syncthreads();

    int buf = 0;
    for (int k0 = 0; k0 < K; k0 += 32) {
        const bool more = (k0 + 32) < K;

        // ---- issue next tile while we compute this one
#if USE_ASYNC
        if (more) {
            #pragma unroll
            for (int it = 0; it < 2; ++it) {
                int r = lr + it * 64;
                __builtin_amdgcn_global_load_async_to_lds_b128(
                    (v4i_*)(A + (size_t)(row0 + r) * K + k0 + 32 + lk),
                    (v4i_*)&Asm[buf ^ 1][r * 40 + lk], 0, 0);
                __builtin_amdgcn_global_load_async_to_lds_b128(
                    (v4i_*)(Wt + (size_t)(n0 + r) * K + k0 + 32 + lk),
                    (v4i_*)&Bsm[buf ^ 1][r * 40 + lk], 0, 0);
            }
        }
#else
        uint4 nar[2], nbr[2];
        if (more) {
            #pragma unroll
            for (int it = 0; it < 2; ++it) {
                int r = lr + it * 64;
                nar[it] = *(const uint4*)(A  + (size_t)(row0 + r) * K + k0 + 32 + lk);
                nbr[it] = *(const uint4*)(Wt + (size_t)(n0 + r) * K + k0 + 32 + lk);
            }
            __builtin_prefetch(A + (size_t)(row0 + lr) * K + k0 + 64, 0, 0);
        }
#endif

        // ---- fragments from LDS (layouts per CDNA5 ISA 7.12.2)
        v16bf af[2], bfr[4];
        #pragma unroll
        for (int mt = 0; mt < 2; ++mt) {
            int rr = wm + mt * 16 + l16;
            ((uint4*)&af[mt])[0] = *(const uint4*)&Asm[buf][rr * 40 + half * 8];       // k = half*8+j
            ((uint4*)&af[mt])[1] = *(const uint4*)&Asm[buf][rr * 40 + 16 + half * 8];  // k = 16+half*8+j
        }
        #pragma unroll
        for (int nt = 0; nt < 4; ++nt) {
            int cc = wn + nt * 16 + l16;
            ((uint4*)&bfr[nt])[0] = *(const uint4*)&Bsm[buf][cc * 40 + half * 8];      // k = half*8+j
            ((uint4*)&bfr[nt])[1] = *(const uint4*)&Bsm[buf][cc * 40 + 16 + half * 8];
        }
        #pragma unroll
        for (int mt = 0; mt < 2; ++mt)
            #pragma unroll
            for (int nt = 0; nt < 4; ++nt)
                acc[mt * 4 + nt] = __builtin_amdgcn_wmma_f32_16x16x32_bf16(
                    false, af[mt], false, bfr[nt], (short)0, acc[mt * 4 + nt], false, false);

        // ---- commit next tile
#if USE_ASYNC
        if (more) __builtin_amdgcn_s_wait_asynccnt(0);
#else
        if (more) {
            #pragma unroll
            for (int it = 0; it < 2; ++it) {
                int r = lr + it * 64;
                *(uint4*)&Asm[buf ^ 1][r * 40 + lk] = nar[it];
                *(uint4*)&Bsm[buf ^ 1][r * 40 + lk] = nbr[it];
            }
        }
#endif
        __syncthreads();
        buf ^= 1;
    }

    // ---- epilogue
    #pragma unroll
    for (int mt = 0; mt < 2; ++mt)
        #pragma unroll
        for (int nt = 0; nt < 4; ++nt) {
            v8f a = acc[mt * 4 + nt];
            #pragma unroll
            for (int r = 0; r < 8; ++r) {
                int row = row0 + wm + mt * 16 + r + 8 * half;
                int col = n0 + wn + nt * 16 + l16;
                float v = a[r];
                if (EP == 0) {
                    ((__bf16*)outv)[(size_t)row * N + col] = (__bf16)v;
                } else if (EP == 1) {
                    float xg = v + bias[col];
                    float t  = tanhf(0.7978845608f * (xg + 0.044715f * xg * xg * xg));
                    ((__bf16*)outv)[(size_t)row * N + col] = (__bf16)(0.5f * xg * (1.f + t));
                } else {
                    ((float*)outv)[(size_t)row * N + col] =
                        v + bias[col] + res[(size_t)row * N + col];
                }
            }
        }
}

// ---------------------------------------------------------------------------
// V [B,S,D] (bf16) -> Vt [B*H, HD, S] (bf16), LDS-tiled 64x64 transpose
// ---------------------------------------------------------------------------
__global__ __launch_bounds__(256) void transpose_v(const __bf16* __restrict__ v,
                                                   __bf16* __restrict__ vt) {
    __shared__ __bf16 tl[64 * 72];
    const int bh = blockIdx.y;
    const int b  = bh >> 4;
    const int h  = bh & 15;
    const int k0 = blockIdx.x * 64;
    const int tid = threadIdx.x;
    #pragma unroll
    for (int it = 0; it < 2; ++it) {
        int r = (tid >> 3) + it * 32;       // key within tile
        int c = (tid & 7) * 8;              // hd seg
        uint4 d = *(const uint4*)(v + (size_t)(b * S_ + k0 + r) * D_ + h * 64 + c);
        *(uint4*)&tl[r * 72 + c] = d;
    }
    __syncthreads();
    {
        int hd   = tid >> 2;
        int kseg = (tid & 3) * 16;
        __attribute__((aligned(16))) __bf16 tmp[16];
        #pragma unroll
        for (int j = 0; j < 16; ++j) tmp[j] = tl[(kseg + j) * 72 + hd];
        __bf16* dst = vt + ((size_t)bh * 64 + hd) * S_ + k0 + kseg;
        *(uint4*)dst       = ((uint4*)tmp)[0];
        *(uint4*)(dst + 8) = ((uint4*)tmp)[1];
    }
}

// ---------------------------------------------------------------------------
// Causal flash attention.  One wave per 16-query tile; 8 waves / block.
// q,k: bf16 [B,S,D];  vt: bf16 [B*H, HD, S];  ctx out: bf16 [B,S,D]
// ---------------------------------------------------------------------------
__global__ __launch_bounds__(256) void attn(const __bf16* __restrict__ q,
                                            const __bf16* __restrict__ k,
                                            const __bf16* __restrict__ vt,
                                            __bf16* __restrict__ ctx) {
    __shared__ __bf16 plds[8][16 * 40];   // per-wave P scratch (16x32, stride 40)
    const int wave = threadIdx.x >> 5;
    const int lane = threadIdx.x & 31;
    const int half = lane >> 4;
    const int l16  = lane & 15;
    const int bh   = blockIdx.x >> 3;
    const int qt   = (blockIdx.x & 7) * 8 + wave;   // query tile 0..63
    const int b    = bh >> 4;
    const int h    = bh & 15;

    const __bf16* qb = q  + (size_t)b * S_ * D_ + h * 64;
    const __bf16* kb = k  + (size_t)b * S_ * D_ + h * 64;
    const __bf16* vb = vt + (size_t)bh * 64 * S_;
    __bf16*       ob = ctx + (size_t)b * S_ * D_ + h * 64;
    __bf16*       pl = &plds[wave][0];

    // Q fragments (A layout), 2 k-chunks of 32 over HD=64
    v16bf qf[2];
    #pragma unroll
    for (int c = 0; c < 2; ++c) {
        const __bf16* qp = qb + (size_t)(qt * 16 + l16) * D_ + c * 32 + half * 8;
        ((uint4*)&qf[c])[0] = *(const uint4*)qp;
        ((uint4*)&qf[c])[1] = *(const uint4*)(qp + 16);
    }

    float mrun[8], lrun[8];
    v8f o[4] = {};
    #pragma unroll
    for (int r = 0; r < 8; ++r) { mrun[r] = -3.0e38f; lrun[r] = 0.f; }

    const int qmax = qt * 16 + 15;
    const int nch  = qmax / 32 + 1;

    for (int kc = 0; kc < nch; ++kc) {
        const int kb0 = kc * 32;
        // ---- scores: two 16x16 tiles (keys kb0..+15, kb0+16..+31)
        v8f st[2] = {};
        #pragma unroll
        for (int t = 0; t < 2; ++t)
            #pragma unroll
            for (int c = 0; c < 2; ++c) {
                v16bf kf;
                const __bf16* kp = kb + (size_t)(kb0 + t * 16 + l16) * D_ + c * 32 + half * 8;
                ((uint4*)&kf)[0] = *(const uint4*)kp;
                ((uint4*)&kf)[1] = *(const uint4*)(kp + 16);
                st[t] = __builtin_amdgcn_wmma_f32_16x16x32_bf16(
                    false, qf[c], false, kf, (short)0, st[t], false, false);
            }
        // ---- online softmax (rows live in 16-lane groups; row = r + 8*half)
        float p0[8], p1[8], fac[8];
        #pragma unroll
        for (int r = 0; r < 8; ++r) {
            int qrow = qt * 16 + r + 8 * half;
            float s0 = st[0][r] * 0.125f;
            float s1 = st[1][r] * 0.125f;
            if (kb0 + l16 > qrow)      s0 = -3.0e38f;
            if (kb0 + 16 + l16 > qrow) s1 = -3.0e38f;
            float rv = fmaxf(s0, s1);
            #pragma unroll
            for (int off = 8; off >= 1; off >>= 1) rv = fmaxf(rv, __shfl_xor(rv, off, 16));
            float mn = fmaxf(mrun[r], rv);
            float f  = __expf(mrun[r] - mn);
            mrun[r] = mn;
            float e0 = __expf(s0 - mn);
            float e1 = __expf(s1 - mn);
            float rs = e0 + e1;
            #pragma unroll
            for (int off = 8; off >= 1; off >>= 1) rs += __shfl_xor(rs, off, 16);
            lrun[r] = lrun[r] * f + rs;
            p0[r] = e0; p1[r] = e1; fac[r] = f;
        }
        #pragma unroll
        for (int t = 0; t < 4; ++t)
            #pragma unroll
            for (int r = 0; r < 8; ++r) o[t][r] *= fac[r];

        // ---- P (C layout) -> LDS -> A-fragment layout
        #pragma unroll
        for (int r = 0; r < 8; ++r) {
            int row = r + 8 * half;
            pl[row * 40 + l16]      = (__bf16)p0[r];
            pl[row * 40 + 16 + l16] = (__bf16)p1[r];
        }
        asm volatile("s_wait_dscnt 0x0" ::: "memory");
        v16bf pf;
        {
            const __bf16* pp = pl + l16 * 40 + half * 8;
            ((uint4*)&pf)[0] = *(const uint4*)pp;
            ((uint4*)&pf)[1] = *(const uint4*)(pp + 16);
        }
        // ---- O += P * V   (B fragments contiguous from Vt)
        #pragma unroll
        for (int t = 0; t < 4; ++t) {
            v16bf vf;
            const __bf16* vp = vb + (size_t)(t * 16 + l16) * S_ + kb0 + half * 8;
            ((uint4*)&vf)[0] = *(const uint4*)vp;
            ((uint4*)&vf)[1] = *(const uint4*)(vp + 16);
            o[t] = __builtin_amdgcn_wmma_f32_16x16x32_bf16(
                false, pf, false, vf, (short)0, o[t], false, false);
        }
    }

    // ---- normalize + store ctx (bf16)
    #pragma unroll
    for (int t = 0; t < 4; ++t)
        #pragma unroll
        for (int r = 0; r < 8; ++r) {
            int row = qt * 16 + r + 8 * half;
            ob[(size_t)row * D_ + t * 16 + l16] = (__bf16)(o[t][r] / lrun[r]);
        }
}

// ---------------------------------------------------------------------------
// Host-side orchestration
// ---------------------------------------------------------------------------
extern "C" void kernel_launch(void* const* d_in, const int* in_sizes, int n_in,
                              void* d_out, int out_size, void* d_ws, size_t ws_size,
                              hipStream_t stream) {
    (void)in_sizes; (void)n_in; (void)out_size; (void)ws_size;
    const float* x     = (const float*)d_in[0];
    const float* WQ    = (const float*)d_in[1];
    const float* WK    = (const float*)d_in[2];
    const float* WV    = (const float*)d_in[3];
    const float* Wo    = (const float*)d_in[4];
    const float* bo    = (const float*)d_in[5];
    const float* W1    = (const float*)d_in[6];
    const float* b1    = (const float*)d_in[7];
    const float* W2    = (const float*)d_in[8];
    const float* b2    = (const float*)d_in[9];
    const float* ln1_s = (const float*)d_in[10];
    const float* ln1_b = (const float*)d_in[11];
    const float* ln2_s = (const float*)d_in[12];
    const float* ln2_b = (const float*)d_in[13];

    char* ws = (char*)d_ws;
    size_t off = 0;
    auto alloc = [&](size_t bytes) -> void* {
        void* p = ws + off;
        off += (bytes + 255) & ~(size_t)255;
        return p;
    };
    __bf16* h_bf   = (__bf16*)alloc((size_t)ROWS_ * D_ * 2);
    __bf16* wqt    = (__bf16*)alloc((size_t)D_ * D_ * 2);      // [N][K]
    __bf16* wkt    = (__bf16*)alloc((size_t)D_ * D_ * 2);
    __bf16* wvt    = (__bf16*)alloc((size_t)D_ * D_ * 2);
    __bf16* wot    = (__bf16*)alloc((size_t)D_ * D_ * 2);
    __bf16* w1t    = (__bf16*)alloc((size_t)D_ * 4 * D_ * 2);  // [4096][1024]
    __bf16* w2t    = (__bf16*)alloc((size_t)4 * D_ * D_ * 2);  // [1024][4096]
    __bf16* q_bf   = (__bf16*)alloc((size_t)ROWS_ * D_ * 2);
    __bf16* k_bf   = (__bf16*)alloc((size_t)ROWS_ * D_ * 2);
    __bf16* v_bf   = (__bf16*)alloc((size_t)ROWS_ * D_ * 2);
    __bf16* vt_bf  = (__bf16*)alloc((size_t)ROWS_ * D_ * 2);
    __bf16* ctx_bf = (__bf16*)alloc((size_t)ROWS_ * D_ * 2);
    float*  x2     = (float*)alloc((size_t)ROWS_ * D_ * 4);
    __bf16* h2_bf  = (__bf16*)alloc((size_t)ROWS_ * D_ * 2);
    __bf16* hff_bf = (__bf16*)alloc((size_t)ROWS_ * 4 * D_ * 2);

    // weights -> bf16, transposed to [N][K]
    cvt_transpose<<<dim3(16, 16), 256, 0, stream>>>(WQ, wqt, D_, D_);
    cvt_transpose<<<dim3(16, 16), 256, 0, stream>>>(WK, wkt, D_, D_);
    cvt_transpose<<<dim3(16, 16), 256, 0, stream>>>(WV, wvt, D_, D_);
    cvt_transpose<<<dim3(16, 16), 256, 0, stream>>>(Wo, wot, D_, D_);
    cvt_transpose<<<dim3(64, 16), 256, 0, stream>>>(W1, w1t, D_, 4 * D_);
    cvt_transpose<<<dim3(16, 64), 256, 0, stream>>>(W2, w2t, 4 * D_, D_);

    ln_bf16<<<ROWS_, 256, 0, stream>>>(x, ln1_s, ln1_b, h_bf);

    dim3 gD(D_ / 128, ROWS_ / 128);          // N=1024 GEMMs
    gemm_bf16<0><<<gD, 256, 0, stream>>>(h_bf, wqt, nullptr, nullptr, q_bf, D_, D_);
    gemm_bf16<0><<<gD, 256, 0, stream>>>(h_bf, wkt, nullptr, nullptr, k_bf, D_, D_);
    gemm_bf16<0><<<gD, 256, 0, stream>>>(h_bf, wvt, nullptr, nullptr, v_bf, D_, D_);

    transpose_v<<<dim3(S_ / 64, B_ * H_), 256, 0, stream>>>(v_bf, vt_bf);
    attn<<<B_ * H_ * 8, 256, 0, stream>>>(q_bf, k_bf, vt_bf, ctx_bf);

    gemm_bf16<2><<<gD, 256, 0, stream>>>(ctx_bf, wot, bo, x, x2, D_, D_);

    ln_bf16<<<ROWS_, 256, 0, stream>>>(x2, ln2_s, ln2_b, h2_bf);

    dim3 gF(4 * D_ / 128, ROWS_ / 128);      // N=4096
    gemm_bf16<1><<<gF, 256, 0, stream>>>(h2_bf, w1t, b1, nullptr, hff_bf, 4 * D_, D_);
    gemm_bf16<2><<<gD, 256, 0, stream>>>(hff_bf, w2t, b2, x2, (float*)d_out, D_, 4 * D_);
}
